// DMTetGeometry_62277025792158
// MI455X (gfx1250) — compile-verified, matrix operand order checked by reading, not run on hardware
//
#include <hip/hip_runtime.h>

// ---------------------------------------------------------------------------
// DMTet: posenc MLP SDF (WMMA f16->f32, async-LDS weight pipeline)
//        + marching tetrahedra (hash dedup)
// Target: gfx1250 (CDNA5, wave32, WMMA 16x16x32 f16, ASYNCcnt staging)
// ---------------------------------------------------------------------------

#define NV 200000
#define NT 800000

typedef _Float16 half_t;
typedef __attribute__((ext_vector_type(16))) _Float16 v16h;
typedef __attribute__((ext_vector_type(8)))  float    v8f;
typedef __attribute__((ext_vector_type(4)))  int      v4i;

#ifndef __has_builtin
#define __has_builtin(x) 0
#endif

#if __has_builtin(__builtin_amdgcn_global_load_async_to_lds_b128)
#define HAS_ASYNC 1
#else
#define HAS_ASYNC 0
#endif

typedef __attribute__((address_space(1))) v4i gv4i;  // global 16-B packet
typedef __attribute__((address_space(3))) v4i lv4i;  // LDS    16-B packet

__device__ __constant__ int c_tri[16][6] = {
    {-1,-1,-1,-1,-1,-1},{1,0,2,-1,-1,-1},{4,0,3,-1,-1,-1},{1,4,2,1,3,4},
    {3,1,5,-1,-1,-1},{2,3,0,2,5,3},{1,4,0,1,5,4},{4,2,5,-1,-1,-1},
    {4,5,2,-1,-1,-1},{4,1,0,4,5,1},{3,2,0,3,5,2},{1,3,5,-1,-1,-1},
    {4,1,2,4,3,1},{3,0,4,-1,-1,-1},{2,0,1,-1,-1,-1},{-1,-1,-1,-1,-1,-1}};
__device__ __constant__ int c_ntri[16] = {0,1,1,2,1,2,2,1,1,2,2,1,2,1,1,0};
__device__ __constant__ int c_be[12]   = {0,1,0,2,0,3,1,2,1,3,2,3};

// ---------------------------------------------------------------------------
// WMMA helpers (16x16x32 f16 -> f32).  Layouts per CDNA5 ISA 7.12.2.
// ---------------------------------------------------------------------------
__device__ inline v16h load_frag(const half_t* __restrict__ base, int stride) {
  const int lane = threadIdx.x & 31;
  const int m  = lane & 15;
  const int hi = (lane >> 4) << 3;  // 0 or 8
  const half_t* row = base + m * stride;
  v16h f;
#pragma unroll
  for (int v = 0; v < 8; ++v) {
    int k = ((v & 4) << 2) + hi + ((v & 3) << 1);
    f[2 * v]     = row[k];
    f[2 * v + 1] = row[k + 1];
  }
  return f;
}

__device__ inline v8f wmma_f16(v16h a, v16h b, v8f c) {
  return __builtin_amdgcn_wmma_f32_16x16x32_f16(
      false, a, false, b, (short)0, c, false, false);
}

// ---------------------------------------------------------------------------
// Async weight staging: each thread copies one 32-half row (64 B) of the
// current K-chunk, global -> LDS, tracked by ASYNCcnt (no VGPR round trip).
// Builtin signature (from probe): (v4i AS1*, v4i AS3*, imm offset, imm cpol)
// ---------------------------------------------------------------------------
__device__ inline void stage_row(const half_t* g, half_t* l) {
#if HAS_ASYNC
  gv4i* gp = (gv4i*)g;
  lv4i* lp = (lv4i*)l;
#pragma unroll
  for (int i = 0; i < 4; ++i)
    __builtin_amdgcn_global_load_async_to_lds_b128(gp + i, lp + i, 0, 0);
#else
  unsigned int* dst = (unsigned int*)l;
  const unsigned int* src = (const unsigned int*)g;
#pragma unroll
  for (int i = 0; i < 16; ++i) dst[i] = src[i];
#endif
}

__device__ inline void wait_async() {
#if HAS_ASYNC
#if __has_builtin(__builtin_amdgcn_s_wait_asynccnt)
  __builtin_amdgcn_s_wait_asynccnt(0);
#else
  asm volatile("s_wait_asynccnt 0x0" ::: "memory");
#endif
#endif
}

// ---------------------------------------------------------------------------
// Workspace zero (hash keys + counters) — every launch
// ---------------------------------------------------------------------------
__global__ void k_zero(unsigned long long* keys, unsigned int* cnt,
                       unsigned int entries) {
  if (blockIdx.x == 0 && threadIdx.x < 2) cnt[threadIdx.x] = 0u;
  for (size_t i = (size_t)blockIdx.x * blockDim.x + threadIdx.x; i < entries;
       i += (size_t)gridDim.x * blockDim.x)
    keys[i] = 0ull;
}

// ---------------------------------------------------------------------------
// Weight conversion: f32 row-major [K][N] -> f16 transposed [N][K]
// ---------------------------------------------------------------------------
__global__ void k_prep(const float* __restrict__ w0, const float* __restrict__ w1,
                       const float* __restrict__ w2, const float* __restrict__ w3,
                       half_t* __restrict__ w0T, half_t* __restrict__ w1T,
                       half_t* __restrict__ w2T, half_t* __restrict__ w3T) {
  int i = blockIdx.x * blockDim.x + threadIdx.x;
  const int TOT = 256 * 32 + 3 * 256 * 256;
  if (i >= TOT) return;
  if (i < 256 * 32) {
    int n = i >> 5, k = i & 31;
    w0T[i] = (k < 27) ? (half_t)w0[k * 256 + n] : (half_t)0.f;
    return;
  }
  i -= 256 * 32;
  int which = i >> 16, j = i & 65535;
  int n = j >> 8, k = j & 255;
  const float* w = (which == 0) ? w1 : (which == 1) ? w2 : w3;
  half_t* o      = (which == 0) ? w1T : (which == 1) ? w2T : w3T;
  o[(size_t)n * 256 + k] = (half_t)w[(size_t)k * 256 + n];
}

// ---------------------------------------------------------------------------
// Positional encoding -> f16 rows [NV][32]
// ---------------------------------------------------------------------------
__global__ void k_posenc(const float* __restrict__ pos, half_t* __restrict__ enc,
                         int n) {
  int v = blockIdx.x * blockDim.x + threadIdx.x;
  if (v >= n) return;
  float x = pos[3 * v + 0], y = pos[3 * v + 1], z = pos[3 * v + 2];
  half_t* r = enc + (size_t)v * 32;
  r[0] = (half_t)x; r[1] = (half_t)y; r[2] = (half_t)z;
  int o = 3;
#pragma unroll
  for (int i = 0; i < 4; ++i) {
    float f = (float)(1 << i) * 3.14159265358979323846f;
    r[o++] = (half_t)__sinf(f * x);
    r[o++] = (half_t)__sinf(f * y);
    r[o++] = (half_t)__sinf(f * z);
    r[o++] = (half_t)__cosf(f * x);
    r[o++] = (half_t)__cosf(f * y);
    r[o++] = (half_t)__cosf(f * z);
  }
#pragma unroll
  for (int j = 27; j < 32; ++j) r[j] = (half_t)0.f;
}

// ---------------------------------------------------------------------------
// Full MLP with WMMA + async double-buffered weight chunks.
// 256 threads (8 wave32), 32 rows/block.  Each wave: 16-row tile x 64 cols.
// ---------------------------------------------------------------------------
struct MlpW {
  const half_t *w0T, *w1T, *w2T, *w3T;
  const float *b0, *b1, *b2, *b3, *wf, *bf;
};

#define WT_STRIDE 48  // halfs: 96 B rows -> 16B-aligned b128 async writes

__global__ __launch_bounds__(256) void k_mlp(const half_t* __restrict__ enc,
                                             MlpW W, float* __restrict__ sdf) {
  __shared__ __align__(16) half_t act[2][32 * 256];          // 32 KB ping-pong
  __shared__ __align__(16) half_t wbuf[2][256 * WT_STRIDE];  // 48 KB ping-pong
  __shared__ float red[256];

  const int tid   = threadIdx.x;
  const int lane  = tid & 31;
  const int wave  = tid >> 5;
  const int rtile = wave >> 2;        // 0..1
  const int cgrp  = (wave & 3) * 64;  // base column of this wave's 4 N-tiles
  const int gbase = blockIdx.x * 32;

  // stage this block's encoded rows into act[0] (row stride 256, K=0..31)
  {
    int row = tid >> 3, c4 = (tid & 7) * 4;
    *(uint2*)&act[0][row * 256 + c4] =
        *(const uint2*)(enc + (size_t)(gbase + row) * 32 + c4);
  }

  int cur = 0;
  for (int layer = 0; layer < 4; ++layer) {
    const half_t* wg = (layer == 0) ? W.w0T
                     : (layer == 1) ? W.w1T
                     : (layer == 2) ? W.w2T : W.w3T;
    const float* bgl = (layer == 0) ? W.b0
                     : (layer == 1) ? W.b1
                     : (layer == 2) ? W.b2 : W.b3;
    const int K   = (layer == 0) ? 32 : 256;
    const int nkt = K >> 5;
    v8f acc[4] = {};

    // prefetch chunk 0 (async, tracked by ASYNCcnt)
    stage_row(wg + (size_t)tid * K, &wbuf[0][tid * WT_STRIDE]);

    for (int kt = 0; kt < nkt; ++kt) {
      wait_async();     // this wave's LDS writes landed
      __syncthreads();  // everyone's chunk visible; prev-buf readers done
      if (kt + 1 < nkt)  // prefetch next chunk into the other buffer
        stage_row(wg + (size_t)tid * K + (kt + 1) * 32,
                  &wbuf[(kt + 1) & 1][tid * WT_STRIDE]);

      v16h a = load_frag(&act[cur][rtile * 16 * 256 + kt * 32], 256);
      const half_t* wb = wbuf[kt & 1];
#pragma unroll
      for (int t = 0; t < 4; ++t) {
        v16h b = load_frag(&wb[(cgrp + t * 16) * WT_STRIDE], WT_STRIDE);
        acc[t] = wmma_f16(a, b, acc[t]);
      }
    }
    __syncthreads();  // all fragment reads done before buffers are reused

    // epilogue: bias + ReLU, pack f16 into the other activation buffer
    const int nxt = cur ^ 1;
    const int n   = lane & 15;
    const int hi8 = (lane >> 4) << 3;
#pragma unroll
    for (int t = 0; t < 4; ++t) {
      int col = cgrp + t * 16 + n;
      float bias = bgl[col];
#pragma unroll
      for (int v = 0; v < 8; ++v) {
        int row = rtile * 16 + hi8 + v;
        float x = acc[t][v] + bias;
        act[nxt][row * 256 + col] = (half_t)fmaxf(x, 0.f);
      }
    }
    cur = nxt;
  }

  // final layer: sdf = h . wf + bf
  __syncthreads();
  {
    int row = tid >> 3, seg = tid & 7;
    const half_t* h = &act[cur][row * 256 + seg * 32];
    float s = 0.f;
#pragma unroll
    for (int k = 0; k < 32; ++k) s += (float)h[k] * W.wf[seg * 32 + k];
    red[tid] = s;
  }
  __syncthreads();
  if (tid < 32) {
    float s = W.bf[0];
#pragma unroll
    for (int i = 0; i < 8; ++i) s += red[tid * 8 + i];
    sdf[gbase + tid] = s;
  }
}

// ---------------------------------------------------------------------------
// Marching tets: hash-dedup crossing edges, interpolate verts, emit faces
// ---------------------------------------------------------------------------
__device__ inline unsigned int hmix(unsigned long long k) {
  k ^= k >> 33; k *= 0xff51afd7ed558ccdull; k ^= k >> 33;
  return (unsigned int)k;
}

__global__ void k_edge_insert(const int* __restrict__ tet,
                              const float* __restrict__ sdf,
                              unsigned long long* __restrict__ keys,
                              unsigned int* __restrict__ vals,
                              uint2* __restrict__ edges,
                              unsigned int* __restrict__ cnt,
                              unsigned int entries) {
  int t = blockIdx.x * blockDim.x + threadIdx.x;
  if (t >= NT) return;
  int vi[4];
#pragma unroll
  for (int j = 0; j < 4; ++j) vi[j] = tet[4 * t + j];
  int occ[4], s = 0;
#pragma unroll
  for (int j = 0; j < 4; ++j) { occ[j] = sdf[vi[j]] > 0.f; s += occ[j]; }
  if (s == 0 || s == 4) return;
  const unsigned int mask = entries - 1u;
#pragma unroll
  for (int e = 0; e < 6; ++e) {
    int ia = c_be[2 * e], ib = c_be[2 * e + 1];
    if (occ[ia] == occ[ib]) continue;
    unsigned int lo = (unsigned int)min(vi[ia], vi[ib]);
    unsigned int hi = (unsigned int)max(vi[ia], vi[ib]);
    unsigned long long key = ((unsigned long long)lo << 32) | hi;
    unsigned int slot = hmix(key) & mask;
    for (int p = 0; p < 4096; ++p) {
      unsigned long long old = atomicCAS(&keys[slot], 0ull, key);
      if (old == 0ull) {
        unsigned int vid = atomicAdd(&cnt[0], 1u);
        if (vid < entries) { vals[slot] = vid; edges[vid] = make_uint2(lo, hi); }
        break;
      }
      if (old == key) break;
      slot = (slot + 1u) & mask;
    }
  }
}

__global__ void k_verts(const uint2* __restrict__ edges,
                        const float* __restrict__ pos,
                        const float* __restrict__ sdf,
                        const unsigned int* __restrict__ cnt,
                        float* __restrict__ out, long out_elems) {
  unsigned int vid = blockIdx.x * blockDim.x + threadIdx.x;
  if (vid >= cnt[0]) return;
  uint2 e = edges[vid];
  float s0 = sdf[e.x], s1 = sdf[e.y];
  float inv = 1.f / (s0 - s1);
#pragma unroll
  for (int c = 0; c < 3; ++c) {
    float p0 = pos[3 * (size_t)e.x + c], p1 = pos[3 * (size_t)e.y + c];
    long idx = (long)vid * 3 + c;
    if (idx < out_elems) out[idx] = (p0 * (-s1) + p1 * s0) * inv;
  }
}

__device__ inline int hash_lookup(const unsigned long long* keys,
                                  const unsigned int* vals,
                                  unsigned long long key, unsigned int mask) {
  unsigned int slot = hmix(key) & mask;
  for (int p = 0; p < 4096; ++p) {
    unsigned long long k = keys[slot];
    if (k == key) return (int)vals[slot];
    if (k == 0ull) return -1;
    slot = (slot + 1u) & mask;
  }
  return -1;
}

__global__ void k_faces(const int* __restrict__ tet,
                        const float* __restrict__ sdf,
                        const unsigned long long* __restrict__ keys,
                        const unsigned int* __restrict__ vals,
                        unsigned int* __restrict__ cnt,
                        float* __restrict__ out, long out_elems,
                        unsigned int entries) {
  int t = blockIdx.x * blockDim.x + threadIdx.x;
  if (t >= NT) return;
  int vi[4];
#pragma unroll
  for (int j = 0; j < 4; ++j) vi[j] = tet[4 * t + j];
  int occ[4], s = 0, tetindex = 0;
#pragma unroll
  for (int j = 0; j < 4; ++j) {
    occ[j] = sdf[vi[j]] > 0.f;
    s += occ[j];
    tetindex |= occ[j] << j;
  }
  if (s == 0 || s == 4) return;
  int ntri = c_ntri[tetindex];
  if (ntri == 0) return;
  const unsigned int mask = entries - 1u;
  int evid[6];
#pragma unroll
  for (int e = 0; e < 6; ++e) {
    int ia = c_be[2 * e], ib = c_be[2 * e + 1];
    if (occ[ia] == occ[ib]) { evid[e] = -1; continue; }
    unsigned int lo = (unsigned int)min(vi[ia], vi[ib]);
    unsigned int hi = (unsigned int)max(vi[ia], vi[ib]);
    evid[e] = hash_lookup(keys, vals,
                          ((unsigned long long)lo << 32) | hi, mask);
  }
  unsigned int base = atomicAdd(&cnt[1], (unsigned int)ntri);
  long V = (long)cnt[0];
  int* iout = (int*)out;
  for (int k = 0; k < ntri; ++k)
    for (int j = 0; j < 3; ++j) {
      long p = 3 * V + (long)(base + k) * 3 + j;
      if (p < out_elems) iout[p] = evid[c_tri[tetindex][3 * k + j]];
    }
}

// ---------------------------------------------------------------------------
// Host launcher
// ---------------------------------------------------------------------------
static inline size_t al256(size_t x) { return (x + 255) & ~(size_t)255; }

extern "C" void kernel_launch(void* const* d_in, const int* in_sizes, int n_in,
                              void* d_out, int out_size, void* d_ws,
                              size_t ws_size, hipStream_t stream) {
  const float* pos = (const float*)d_in[0];
  const int*   tet = (const int*)d_in[1];
  const float* w0  = (const float*)d_in[2];
  const float* b0  = (const float*)d_in[3];
  const float* w1  = (const float*)d_in[4];
  const float* b1  = (const float*)d_in[5];
  const float* w2  = (const float*)d_in[6];
  const float* b2  = (const float*)d_in[7];
  const float* w3  = (const float*)d_in[8];
  const float* b3  = (const float*)d_in[9];
  const float* wf  = (const float*)d_in[10];
  const float* bf  = (const float*)d_in[11];
  float* out = (float*)d_out;

  char* ws = (char*)d_ws;
  size_t off = 0;
  size_t off_cnt = off;  off = al256(off + 64);
  size_t off_w0T = off;  off = al256(off + (size_t)256 * 32 * 2);
  size_t off_w1T = off;  off = al256(off + (size_t)256 * 256 * 2);
  size_t off_w2T = off;  off = al256(off + (size_t)256 * 256 * 2);
  size_t off_w3T = off;  off = al256(off + (size_t)256 * 256 * 2);
  size_t off_sdf = off;  off = al256(off + (size_t)NV * 4);
  size_t off_enc = off;  off = al256(off + (size_t)NV * 32 * 2);
  size_t fixed = off;

  size_t avail = (ws_size > fixed) ? (ws_size - fixed) : 0;
  unsigned int entries = 1u << 14;
  while ((size_t)entries * 2u * 20u <= avail && entries < (1u << 24))
    entries <<= 1;
  size_t off_keys = fixed;
  size_t off_vals = al256(off_keys + (size_t)entries * 8);
  size_t off_edge = al256(off_vals + (size_t)entries * 4);

  unsigned int*       cnt   = (unsigned int*)(ws + off_cnt);
  half_t*             w0T   = (half_t*)(ws + off_w0T);
  half_t*             w1T   = (half_t*)(ws + off_w1T);
  half_t*             w2T   = (half_t*)(ws + off_w2T);
  half_t*             w3T   = (half_t*)(ws + off_w3T);
  float*              sdf   = (float*)(ws + off_sdf);
  half_t*             enc   = (half_t*)(ws + off_enc);
  unsigned long long* keys  = (unsigned long long*)(ws + off_keys);
  unsigned int*       vals  = (unsigned int*)(ws + off_vals);
  uint2*              edges = (uint2*)(ws + off_edge);

  k_zero<<<1024, 256, 0, stream>>>(keys, cnt, entries);

  {
    int tot = 256 * 32 + 3 * 256 * 256;
    k_prep<<<(tot + 255) / 256, 256, 0, stream>>>(w0, w1, w2, w3,
                                                  w0T, w1T, w2T, w3T);
  }

  k_posenc<<<(NV + 255) / 256, 256, 0, stream>>>(pos, enc, NV);

  {
    MlpW W{w0T, w1T, w2T, w3T, b0, b1, b2, b3, wf, bf};
    k_mlp<<<NV / 32, 256, 0, stream>>>(enc, W, sdf);
  }

  k_edge_insert<<<(NT + 255) / 256, 256, 0, stream>>>(tet, sdf, keys, vals,
                                                      edges, cnt, entries);
  k_verts<<<(entries + 255) / 256, 256, 0, stream>>>(edges, pos, sdf, cnt,
                                                     out, (long)out_size);
  k_faces<<<(NT + 255) / 256, 256, 0, stream>>>(tet, sdf, keys, vals, cnt,
                                                out, (long)out_size, entries);
}
